// GNNModel_38431367365106
// MI455X (gfx1250) — matrix operand order repaired and checked by reading, not compiled, not run on hardware
//
#include <hip/hip_runtime.h>
#include <hip/hip_bf16.h>
#include <math.h>

// ---------------- problem constants ----------------
#define NB   1024      // batch
#define TT   32        // timesteps
#define FF   10        // input features
#define DD   128       // hidden
#define GG   512       // 4*D gates
#define RR   59        // relations
#define KK   16        // neighbors
#define SD   (2*DD+RR) // W_state row stride = 315

typedef __attribute__((ext_vector_type(16))) _Float16 v16h;
typedef __attribute__((ext_vector_type(8)))  float    v8f;

union Frag16 { v16h h; unsigned u[8]; _Float16 e[16]; };

__device__ __forceinline__ float sigf(float x) { return 1.f / (1.f + __expf(-x)); }
// branchless tanh: monotone, saturates to +/-1, no exec-mask divergence
__device__ __forceinline__ float tanh_fast(float x) {
  const float e = __expf(2.f * x);
  return 1.f - 2.f / (e + 1.f);
}

// ---------------- kernel 1: weight/input prep ----------------
// Wk -> f16 N-major padded [512][32]; Wr -> f16 N-major [512][128];
// windows -> f16 zero-padded [1024][32][32]; feats row 0 = zeros.
__global__ void prep_kernel(const float* __restrict__ Wk, const float* __restrict__ Wr,
                            const float* __restrict__ windows,
                            _Float16* __restrict__ WkT2, _Float16* __restrict__ WrT,
                            _Float16* __restrict__ winPad, float* __restrict__ feats) {
  int idx = blockIdx.x * blockDim.x + threadIdx.x;
  if (idx < GG * 32) {                           // WkT2[n][k], K padded 10 -> 32
    int n = idx >> 5, k = idx & 31;
    WkT2[idx] = (k < FF) ? (_Float16)Wk[k * GG + n] : (_Float16)0.f;
  } else if ((idx -= GG * 32) < GG * DD) {       // WrT[n][k]
    int n = idx >> 7, k = idx & 127;
    WrT[idx] = (_Float16)Wr[k * GG + n];
  } else if ((idx -= GG * DD) < NB * TT * 32) {  // winPad[n][t][k], K padded 10 -> 32
    int nt = idx >> 5, k = idx & 31;             // nt = n*32 + t
    winPad[idx] = (k < FF) ? (_Float16)windows[nt * FF + k] : (_Float16)0.f;
  } else if ((idx -= NB * TT * 32) < DD) {       // feats row 0 = zeros
    feats[idx] = 0.f;
  }
}

// ---------------- kernel 2: LSTM scan, WMMA recurrent matmul ----------------
// grid 64 (16-row tiles), block 256 (8 wave32s, each owning 64 gate columns)
__global__ __launch_bounds__(256) void lstm_kernel(
    const _Float16* __restrict__ winPad, const _Float16* __restrict__ WkT2,
    const _Float16* __restrict__ WrT, const float* __restrict__ bias,
    float* __restrict__ feats) {
  __shared__ float    z_lds[16 * GG];   // 32 KB
  __shared__ _Float16 h_lds[16 * DD];   // 4 KB

  const int tid   = threadIdx.x;
  const int wave  = tid >> 5;
  const int lane  = tid & 31;
  const int m16   = lane & 15;
  const int khalf = (lane >> 4) & 1;
  const int row0  = blockIdx.x * 16;
  const int colbase = wave * 64;

  for (int j = tid; j < 16 * DD; j += 256) h_lds[j] = (_Float16)0.f;

  float c_reg[8];
#pragma unroll
  for (int j = 0; j < 8; ++j) c_reg[j] = 0.f;

  // Loop-invariant B fragments held in VGPRs across the whole scan.
  Frag16 bWk[4];          // windows-path B (K padded to 32)
  Frag16 bWr[4][4];       // [ktile][ntile] recurrent B
  float  bv[4];
#pragma unroll
  for (int nt = 0; nt < 4; ++nt) {
    const int ncol = colbase + nt * 16 + m16;
    bv[nt] = bias[ncol];
#pragma unroll
    for (int v = 0; v < 8; ++v) {
      const int kk = 2 * (v & 3) + 8 * khalf + 16 * (v >> 2);
      bWk[nt].u[v] = *(const unsigned*)(WkT2 + ncol * 32 + kk);
    }
#pragma unroll
    for (int kt = 0; kt < 4; ++kt)
#pragma unroll
      for (int v = 0; v < 8; ++v) {
        const int kk = kt * 32 + 2 * (v & 3) + 8 * khalf + 16 * (v >> 2);
        bWr[kt][nt].u[v] = *(const unsigned*)(WrT + ncol * DD + kk);
      }
  }

  // Window A fragments: branchless aligned b32 loads from the padded copy.
  const _Float16* wp = winPad + (row0 + m16) * (TT * 32);
  Frag16 aw;
#pragma unroll
  for (int v = 0; v < 8; ++v) {
    const int kk = 2 * (v & 3) + 8 * khalf + 16 * (v >> 2);
    aw.u[v] = *(const unsigned*)(wp + kk);          // t = 0
  }
  __syncthreads();

  for (int t = 0; t < TT; ++t) {
    // ---- phase A: z = b + h @ Wr + win_t @ Wk (per-wave 16x64 slice) ----
    v8f acc[4];
#pragma unroll
    for (int nt = 0; nt < 4; ++nt) {
      v8f cinit;
#pragma unroll
      for (int r = 0; r < 8; ++r) cinit[r] = bv[nt];
      acc[nt] = cinit;
    }
    // recurrent term first (LDS-fed) — hides the global window-fragment latency
#pragma unroll
    for (int kt = 0; kt < 4; ++kt) {
      Frag16 ah;
#pragma unroll
      for (int v = 0; v < 8; ++v) {
        const int kk = kt * 32 + 2 * (v & 3) + 8 * khalf + 16 * (v >> 2);
        ah.u[v] = *(const unsigned*)&h_lds[m16 * DD + kk];
      }
#pragma unroll
      for (int nt = 0; nt < 4; ++nt)
        acc[nt] = __builtin_amdgcn_wmma_f32_16x16x32_f16(
            false, ah.h, false, bWr[kt][nt].h, (short)0, acc[nt], false, false);
    }
    // input term
#pragma unroll
    for (int nt = 0; nt < 4; ++nt)
      acc[nt] = __builtin_amdgcn_wmma_f32_16x16x32_f16(
          false, aw.h, false, bWk[nt].h, (short)0, acc[nt], false, false);

    // software-pipeline: issue next step's window fragment loads now; they
    // overlap the z store, barrier and gate phase below.
    if (t + 1 < TT) {
#pragma unroll
      for (int v = 0; v < 8; ++v) {
        const int kk = 2 * (v & 3) + 8 * khalf + 16 * (v >> 2);
        aw.u[v] = *(const unsigned*)(wp + (t + 1) * 32 + kk);
      }
    }

#pragma unroll
    for (int nt = 0; nt < 4; ++nt)
#pragma unroll
      for (int r = 0; r < 8; ++r)
        z_lds[(r + 8 * khalf) * GG + colbase + nt * 16 + m16] = acc[nt][r];
    __syncthreads();

    // ---- gate phase: elementwise LSTM cell update (8 cells / thread) ----
    const bool last = (t == TT - 1);
#pragma unroll
    for (int j = 0; j < 8; ++j) {
      const int flat = tid * 8 + j;
      const int m = flat >> 7, d = flat & 127;
      const float zi = z_lds[m * GG + d];
      const float zf = z_lds[m * GG + DD + d];
      const float zg = z_lds[m * GG + 2 * DD + d];
      const float zo = z_lds[m * GG + 3 * DD + d];
      const float c  = sigf(zf) * c_reg[j] + sigf(zi) * tanh_fast(zg);
      c_reg[j] = c;
      const float h = sigf(zo) * tanh_fast(c);
      h_lds[m * DD + d] = (_Float16)h;
      if (last) feats[(1 + row0 + m) * DD + d] = h;
    }
    __syncthreads();
  }
}

// ---------------- kernel 3: projections (all L2-resident dots) ----------------
__global__ void proj_kernel(const float* __restrict__ feats, const float* __restrict__ W_state,
                            const float* __restrict__ W_relattn,
                            float* __restrict__ cur_term, float* __restrict__ nbr_proj,
                            float* __restrict__ fwn, float* __restrict__ cwc) {
  int idx = blockIdx.x * blockDim.x + threadIdx.x;
  const float* cur = feats + DD; // cur[n] = feats[1+n]
  if (idx < NB * RR) {                                   // cur @ Wc.T
    const int n = idx / RR, r = idx % RR;
    const float* x = cur + n * DD;
    const float* w = W_state + r * SD;
    float s = 0.f;
    for (int d = 0; d < DD; ++d) s += x[d] * w[d];
    cur_term[idx] = s;
  } else if ((idx -= NB * RR) < (NB + 1) * RR) {         // feats @ Wn.T
    const int f = idx / RR, r = idx % RR;
    const float* x = feats + f * DD;
    const float* w = W_state + r * SD + DD;
    float s = 0.f;
    for (int d = 0; d < DD; ++d) s += x[d] * w[d];
    nbr_proj[idx] = s;
  } else if ((idx -= (NB + 1) * RR) < (NB + 1)) {        // feats @ wn
    const float* x = feats + idx * DD;
    const float* w = W_relattn + DD;
    float s = 0.f;
    for (int d = 0; d < DD; ++d) s += x[d] * w[d];
    fwn[idx] = s;
  } else if ((idx -= (NB + 1)) < NB) {                   // cur @ wc
    const float* x = cur + idx * DD;
    float s = 0.f;
    for (int d = 0; d < DD; ++d) s += x[d] * W_relattn[d];
    cwc[idx] = s;
  }
}

// ---------------- kernel 4: s2[r][n] via analytic rel_reps . wn --------------
__global__ void s2_kernel(const int* __restrict__ neighbors, const float* __restrict__ W_state,
                          const float* __restrict__ b_state, const float* __restrict__ W_relattn,
                          const float* __restrict__ b_relattn,
                          const float* __restrict__ cur_term, const float* __restrict__ nbr_proj,
                          const float* __restrict__ fwn, const float* __restrict__ cwc,
                          float* __restrict__ s2) {
  const int idx = blockIdx.x * blockDim.x + threadIdx.x;
  if (idx >= RR * NB) return;
  const int r = idx / NB, n = idx % NB;
  const float rel = W_state[r * SD + 2 * DD + r] + b_state[r];
  const float ct  = cur_term[n * RR + r];
  const int* nb = neighbors + (r * NB + n) * KK;
  float sc[KK]; int id[KK];
  float mx = -1e30f;
#pragma unroll
  for (int k = 0; k < KK; ++k) {
    const int ix = nb[k]; id[k] = ix;
    const float s = fmaxf(ct + nbr_proj[ix * RR + r] + rel, 0.f);
    sc[k] = s; mx = fmaxf(mx, s);
  }
  float sum = 0.f, dot = 0.f;
#pragma unroll
  for (int k = 0; k < KK; ++k) {
    const float e = __expf(sc[k] - mx);
    sum += e; dot += e * fwn[id[k]];
  }
  const float v = cwc[n] + dot / sum + W_relattn[2 * DD + r] + b_relattn[0];
  s2[idx] = fmaxf(v, 0.f);
}

// ---------------- kernel 5: aw softmax, gather-aggregate, head --------------
__global__ __launch_bounds__(64) void final_kernel(
    const int* __restrict__ neighbors, const float* __restrict__ W_state,
    const float* __restrict__ b_state, const float* __restrict__ feats,
    const float* __restrict__ cur_term, const float* __restrict__ nbr_proj,
    const float* __restrict__ s2, const float* __restrict__ W_pred,
    const float* __restrict__ b_pred, float* __restrict__ out) {
  __shared__ float coeff[RR * KK];
  __shared__ int   idxs[RR * KK];
  __shared__ float s2v[64];
  __shared__ float upd[DD];
  __shared__ float lgt[4];
  const int n = blockIdx.x, tid = threadIdx.x;
  const float* cur = feats + DD;

  if (tid < RR) {
    const int r = tid;
    s2v[r] = s2[r * NB + n];
    const float rel = W_state[r * SD + 2 * DD + r] + b_state[r];
    const float ct  = cur_term[n * RR + r];
    const int* nb = neighbors + (r * NB + n) * KK;
    float sc[KK]; int id[KK];
    float mx = -1e30f;
#pragma unroll
    for (int k = 0; k < KK; ++k) {
      const int ix = nb[k]; id[k] = ix;
      const float s = fmaxf(ct + nbr_proj[ix * RR + r] + rel, 0.f);
      sc[k] = s; mx = fmaxf(mx, s);
    }
    float sum = 0.f;
#pragma unroll
    for (int k = 0; k < KK; ++k) { sc[k] = __expf(sc[k] - mx); sum += sc[k]; }
    const float inv = 1.f / sum;
#pragma unroll
    for (int k = 0; k < KK; ++k) { coeff[r * KK + k] = sc[k] * inv; idxs[r * KK + k] = id[k]; }
  }
  __syncthreads();

  // softmax over R (scalar, redundantly per-thread from LDS)
  float mx = -1e30f;
  for (int r = 0; r < RR; ++r) mx = fmaxf(mx, s2v[r]);
  float asum = 0.f;
  for (int r = 0; r < RR; ++r) asum += __expf(s2v[r] - mx);
  const float ainv = 1.f / asum;
  if (tid < RR) {
    const float aw = __expf(s2v[tid] - mx) * ainv;
#pragma unroll
    for (int k = 0; k < KK; ++k) coeff[tid * KK + k] *= aw;
  }
  __syncthreads();

  // rel_agg: each thread owns d = tid and d = tid+64 (coalesced gathers)
  float a0 = 0.f, a1 = 0.f;
  for (int e = 0; e < RR * KK; ++e) {
    const float c = coeff[e];
    const float* f = feats + idxs[e] * DD;
    a0 += c * f[tid];
    a1 += c * f[tid + 64];
  }
  const float u0 = cur[n * DD + tid] + a0;
  const float u1 = cur[n * DD + tid + 64] + a1;
  out[2 * 3 * NB + n * DD + tid]      = u0;
  out[2 * 3 * NB + n * DD + tid + 64] = u1;
  upd[tid] = u0; upd[tid + 64] = u1;
  __syncthreads();

  if (tid < 3) {
    float s = b_pred[tid];
    for (int d = 0; d < DD; ++d) s += upd[d] * W_pred[d * 3 + tid];
    lgt[tid] = s;
  }
  __syncthreads();
  if (tid == 0) {
    const float l0 = lgt[0], l1 = lgt[1], l2 = lgt[2];
    out[n * 3 + 0] = l0; out[n * 3 + 1] = l1; out[n * 3 + 2] = l2;
    const float m = fmaxf(l0, fmaxf(l1, l2));
    const float e0 = __expf(l0 - m), e1 = __expf(l1 - m), e2 = __expf(l2 - m);
    const float inv = 1.f / (e0 + e1 + e2);
    out[3 * NB + n * 3 + 0] = e0 * inv;
    out[3 * NB + n * 3 + 1] = e1 * inv;
    out[3 * NB + n * 3 + 2] = e2 * inv;
  }
}

// ---------------- launcher ----------------
extern "C" void kernel_launch(void* const* d_in, const int* in_sizes, int n_in,
                              void* d_out, int out_size, void* d_ws, size_t ws_size,
                              hipStream_t stream) {
  const float* windows   = (const float*)d_in[0];
  const int*   neighbors = (const int*)  d_in[1];
  const float* Wk        = (const float*)d_in[2];
  const float* Wr        = (const float*)d_in[3];
  const float* bias      = (const float*)d_in[4];
  const float* W_state   = (const float*)d_in[5];
  const float* b_state   = (const float*)d_in[6];
  const float* W_relattn = (const float*)d_in[7];
  const float* b_relattn = (const float*)d_in[8];
  const float* W_pred    = (const float*)d_in[9];
  const float* b_pred    = (const float*)d_in[10];
  float* out = (float*)d_out;

  char* ws = (char*)d_ws;
  _Float16* WkT2     = (_Float16*)(ws + 0);        //  32768 B
  _Float16* WrT      = (_Float16*)(ws + 32768);    // 131072 B
  float*    feats    = (float*)(ws + 163840);      // 1025*128*4 = 524800 B
  float*    cur_term = (float*)(ws + 688640);      // 1024*59*4  = 241664 B
  float*    nbr_proj = (float*)(ws + 930304);      // 1025*59*4  = 241900 B
  float*    fwn      = (float*)(ws + 1172224);     // 1025*4
  float*    cwc      = (float*)(ws + 1176576);     // 1024*4
  float*    s2buf    = (float*)(ws + 1180672);     // 59*1024*4  = 241664 B
  _Float16* winPad   = (_Float16*)(ws + 1422336);  // 1024*32*32*2 = 2097152 B

  {
    const int total = GG * 32 + GG * DD + NB * TT * 32 + DD; // 1,130,624
    prep_kernel<<<(total + 255) / 256, 256, 0, stream>>>(Wk, Wr, windows,
                                                         WkT2, WrT, winPad, feats);
  }
  lstm_kernel<<<NB / 16, 256, 0, stream>>>(winPad, WkT2, WrT, bias, feats);
  {
    const int total = NB * RR + (NB + 1) * RR + (NB + 1) + NB; // 122940
    proj_kernel<<<(total + 255) / 256, 256, 0, stream>>>(feats, W_state, W_relattn,
                                                         cur_term, nbr_proj, fwn, cwc);
  }
  s2_kernel<<<(RR * NB + 255) / 256, 256, 0, stream>>>(neighbors, W_state, b_state,
                                                       W_relattn, b_relattn, cur_term,
                                                       nbr_proj, fwn, cwc, s2buf);
  final_kernel<<<NB, 64, 0, stream>>>(neighbors, W_state, b_state, feats, cur_term,
                                      nbr_proj, s2buf, W_pred, b_pred, out);
}
